// WeightedMultiHeadGAT_58128087384310
// MI455X (gfx1250) — matrix operand order repaired and checked by reading, not compiled, not run on hardware
//
#include <hip/hip_runtime.h>
#include <hip/hip_bf16.h>
#include <math.h>

// Problem constants (match reference)
#define NN    3072
#define FIN   256
#define FOUT  256
#define NH    8
#define DDIM  32
#define ALPHA 0.2f
#define LN_EPS 1e-5f

typedef __attribute__((ext_vector_type(16))) __bf16 v16bf;
typedef __attribute__((ext_vector_type(8)))  __bf16 v8bf;
typedef __attribute__((ext_vector_type(8)))  float  v8f;

// ---------------------------------------------------------------------------
// WMMA fragment helpers (layouts per CDNA5 ISA 7.12.2, wave32)
// A (16x32 bf16, row-major source): lane l (0..15)/hi half holds row M=l,
//   elements i=0..7  -> K = hi*8 + i
//   elements i=8..15 -> K = 16 + hi*8 + (i-8)
// B (32x16 bf16) with B[k][n] = S[n][k] (S row-major): same K mapping as A,
//   N = lane&15  => two 8-element (16B) loads per lane.
// C/D (16x16 f32): VGPR r: M = r + 8*hi, N = lane&15.
// ---------------------------------------------------------------------------
static __device__ __forceinline__ v16bf load_a_frag(const __bf16* base, int stride,
                                                    int m0, int k0, int l, int hi) {
    const __bf16* row = base + (size_t)(m0 + l) * stride + k0 + hi * 8;
    v8bf lo = *reinterpret_cast<const v8bf*>(row);
    v8bf hv = *reinterpret_cast<const v8bf*>(row + 16);
    return __builtin_shufflevector(lo, hv, 0,1,2,3,4,5,6,7,8,9,10,11,12,13,14,15);
}

static __device__ __forceinline__ v16bf load_bT_frag(const __bf16* base, int stride,
                                                     int n0, int k0, int l, int hi) {
    const __bf16* row = base + (size_t)(n0 + l) * stride + k0 + hi * 8;
    v8bf lo = *reinterpret_cast<const v8bf*>(row);
    v8bf hv = *reinterpret_cast<const v8bf*>(row + 16);
    return __builtin_shufflevector(lo, hv, 0,1,2,3,4,5,6,7,8,9,10,11,12,13,14,15);
}

static __device__ __forceinline__ v8f wmma_bf16(v16bf a, v16bf b, v8f c) {
    return __builtin_amdgcn_wmma_f32_16x16x32_bf16(false, a, false, b, (short)0, c,
                                                   false, false);
}

// ---------------------------------------------------------------------------
// K1: fp32 -> bf16 conversion
// ---------------------------------------------------------------------------
__global__ void k_cvt_bf16(const float* __restrict__ src, __bf16* __restrict__ dst, int n) {
    int i = blockIdx.x * blockDim.x + threadIdx.x;
    if (i < n) dst[i] = (__bf16)src[i];
}

// ---------------------------------------------------------------------------
// K2: h = x @ W^T  (3072x256x256), write h (f32) and hT (bf16 transposed [FOUT][NN])
// 8 waves / WG; wave w owns output columns [32w, 32w+32); WG owns 16 rows.
// ---------------------------------------------------------------------------
__global__ void k_gemm_xw(const __bf16* __restrict__ xb, const __bf16* __restrict__ Wb,
                          float* __restrict__ h, __bf16* __restrict__ hT) {
    const int lane = threadIdx.x & 31;
    const int wave = threadIdx.x >> 5;
    const int l = lane & 15, hi = lane >> 4;
    const int mb = blockIdx.x * 16;
    const int nb = wave * 32;

    v8f c0 = {0,0,0,0,0,0,0,0};
    v8f c1 = {0,0,0,0,0,0,0,0};
    for (int kb = 0; kb < FIN; kb += 32) {
        v16bf a  = load_a_frag(xb, FIN, mb, kb, l, hi);
        v16bf b0 = load_bT_frag(Wb, FIN, nb,      kb, l, hi);
        v16bf b1 = load_bT_frag(Wb, FIN, nb + 16, kb, l, hi);
        c0 = wmma_bf16(a, b0, c0);
        c1 = wmma_bf16(a, b1, c1);
    }
#pragma unroll
    for (int r = 0; r < 8; ++r) {
        int m = mb + r + 8 * hi;
        int n0 = nb + l, n1 = nb + 16 + l;
        h[(size_t)m * FOUT + n0] = c0[r];
        h[(size_t)m * FOUT + n1] = c1[r];
        hT[(size_t)n0 * NN + m] = (__bf16)c0[r];
        hT[(size_t)n1 * NN + m] = (__bf16)c1[r];
    }
}

// ---------------------------------------------------------------------------
// K3: a_src/a_dst, stored head-major: asrcT[h][n], adstT[h][n]
// ---------------------------------------------------------------------------
__global__ void k_attn_vec(const float* __restrict__ h, const float* __restrict__ attn_src,
                           const float* __restrict__ attn_dst,
                           float* __restrict__ asrcT, float* __restrict__ adstT) {
    int t = blockIdx.x * blockDim.x + threadIdx.x;
    if (t >= NN * NH) return;
    int n = t >> 3, hh = t & 7;
    const float* hr = h + (size_t)n * FOUT + hh * DDIM;
    float s = 0.f, d = 0.f;
#pragma unroll
    for (int k = 0; k < DDIM; ++k) {
        float v = hr[k];
        s += v * attn_src[hh * DDIM + k];
        d += v * attn_dst[hh * DDIM + k];
    }
    asrcT[(size_t)hh * NN + n] = s;
    adstT[(size_t)hh * NN + n] = d;
}

// ---------------------------------------------------------------------------
// K4: fused attention (mask + leaky-relu + softmax over j) + aggregation WMMA.
// WG = 16 output rows x 8 heads; wave w handles head w.
// Pass1: online softmax stats (m, s) per row; parked in lane (lane&15)==row.
// Pass2: P built directly in A-fragment layout; 2 WMMAs per 32-j chunk vs hT.
// ---------------------------------------------------------------------------
__global__ void k_attn_agg(const float* __restrict__ adj,
                           const float* __restrict__ asrcT, const float* __restrict__ adstT,
                           const float* __restrict__ edge_W, const float* __restrict__ edge_b,
                           const __bf16* __restrict__ hT, __bf16* __restrict__ hout_bf) {
    const int lane = threadIdx.x & 31;
    const int hh   = threadIdx.x >> 5;       // head = wave id
    const int l = lane & 15, hi = lane >> 4;
    const int ib = blockIdx.x * 16;

    const float ew = edge_W[hh];             // edge_W shape (H,1)
    const float eb = edge_b[hh];
    const float* adstH = adstT + (size_t)hh * NN;
    const float* asrcH = asrcT + (size_t)hh * NN;

    float m_mine = 0.f, sinv_mine = 0.f;

    // ---- Pass 1: softmax stats per row ----
    for (int r = 0; r < 16; ++r) {
        const float* adjrow = adj + (size_t)(ib + r) * NN;
        const float asrc_r = asrcH[ib + r];
        float m = -INFINITY, s = 0.f;
        for (int j = lane; j < NN; j += 32) {
            float w = adjrow[j];
            if (w > 0.f) {
                float e = asrc_r + adstH[j] + w * ew + eb;
                e = (e >= 0.f) ? e : ALPHA * e;
                if (e > m) { s = s * __expf(m - e) + 1.0f; m = e; }
                else       { s += __expf(e - m); }
            }
        }
#pragma unroll
        for (int off = 16; off > 0; off >>= 1) {
            float mo = __shfl_xor(m, off, 32);
            float so = __shfl_xor(s, off, 32);
            if (mo > m)            { s = s * __expf(m - mo) + so; m = mo; }
            else if (mo > -INFINITY) s += so * __expf(mo - m);
        }
        if ((lane & 15) == r) {
            m_mine   = m;
            sinv_mine = (s > 0.f) ? (1.0f / s) : 0.f;
        }
    }

    // ---- Pass 2: build P in A-fragment layout, WMMA-accumulate against hT ----
    const float  asrc_mine  = asrcH[ib + l];
    const float* adjrow_mine = adj + (size_t)(ib + l) * NN;
    const __bf16* hTh0 = hT + (size_t)(hh * DDIM + l)      * NN;  // N-tile 0
    const __bf16* hTh1 = hT + (size_t)(hh * DDIM + 16 + l) * NN;  // N-tile 1

    v8f acc0 = {0,0,0,0,0,0,0,0};
    v8f acc1 = {0,0,0,0,0,0,0,0};
    for (int jb = 0; jb < NN; jb += 32) {
        if (jb + 256 < NN) __builtin_prefetch(adjrow_mine + jb + 256, 0, 3);
        v16bf af;
#pragma unroll
        for (int i = 0; i < 16; ++i) {
            int k = (i < 8) ? (hi * 8 + i) : (16 + hi * 8 + (i - 8));
            int j = jb + k;
            float w = adjrow_mine[j];
            float p = 0.f;
            if (w > 0.f) {
                float e = asrc_mine + adstH[j] + w * ew + eb;
                e = (e >= 0.f) ? e : ALPHA * e;
                p = __expf(e - m_mine) * sinv_mine;
            }
            af[i] = (__bf16)p;
        }
        v16bf b0 = load_bT_frag(hTh0, 0, 0, jb, 0, hi);
        v16bf b1 = load_bT_frag(hTh1, 0, 0, jb, 0, hi);
        acc0 = wmma_bf16(af, b0, acc0);
        acc1 = wmma_bf16(af, b1, acc1);
    }
#pragma unroll
    for (int r = 0; r < 8; ++r) {
        int m = ib + r + 8 * hi;
        hout_bf[(size_t)m * FOUT + hh * DDIM + l]      = (__bf16)acc0[r];
        hout_bf[(size_t)m * FOUT + hh * DDIM + 16 + l] = (__bf16)acc1[r];
    }
}

// ---------------------------------------------------------------------------
// K5: y = h_out @ fuse_W^T + b, then LayerNorm + ReLU (fused via LDS tile).
// ---------------------------------------------------------------------------
__global__ void k_gemm_fuse_ln(const __bf16* __restrict__ houtb, const __bf16* __restrict__ fWb,
                               const float* __restrict__ fuse_b,
                               const float* __restrict__ ln_g, const float* __restrict__ ln_b,
                               float* __restrict__ out) {
    __shared__ float ytile[16][FOUT];   // 16 KB
    const int lane = threadIdx.x & 31;
    const int wave = threadIdx.x >> 5;
    const int l = lane & 15, hi = lane >> 4;
    const int mb = blockIdx.x * 16;
    const int nb = wave * 32;

    v8f c0 = {0,0,0,0,0,0,0,0};
    v8f c1 = {0,0,0,0,0,0,0,0};
    for (int kb = 0; kb < FOUT; kb += 32) {
        v16bf a  = load_a_frag(houtb, FOUT, mb, kb, l, hi);
        v16bf b0 = load_bT_frag(fWb, FOUT, nb,      kb, l, hi);
        v16bf b1 = load_bT_frag(fWb, FOUT, nb + 16, kb, l, hi);
        c0 = wmma_bf16(a, b0, c0);
        c1 = wmma_bf16(a, b1, c1);
    }
#pragma unroll
    for (int r = 0; r < 8; ++r) {
        int m = r + 8 * hi;
        ytile[m][nb + l]      = c0[r] + fuse_b[nb + l];
        ytile[m][nb + 16 + l] = c1[r] + fuse_b[nb + 16 + l];
    }
    __syncthreads();

    // LayerNorm + ReLU: wave w normalizes rows 2w, 2w+1
#pragma unroll
    for (int rr = 0; rr < 2; ++rr) {
        int row = wave * 2 + rr;
        float sum = 0.f, sq = 0.f;
        for (int c = lane; c < FOUT; c += 32) {
            float v = ytile[row][c];
            sum += v; sq += v * v;
        }
#pragma unroll
        for (int off = 16; off > 0; off >>= 1) {
            sum += __shfl_xor(sum, off, 32);
            sq  += __shfl_xor(sq,  off, 32);
        }
        float mean = sum * (1.0f / FOUT);
        float var  = sq * (1.0f / FOUT) - mean * mean;
        float rstd = rsqrtf(var + LN_EPS);
        for (int c = lane; c < FOUT; c += 32) {
            float v = (ytile[row][c] - mean) * rstd * ln_g[c] + ln_b[c];
            out[(size_t)(mb + row) * FOUT + c] = fmaxf(v, 0.f);
        }
    }
}

// ---------------------------------------------------------------------------
// Host launcher
// ---------------------------------------------------------------------------
extern "C" void kernel_launch(void* const* d_in, const int* in_sizes, int n_in,
                              void* d_out, int out_size, void* d_ws, size_t ws_size,
                              hipStream_t stream) {
    const float* x        = (const float*)d_in[0];   // (N, FIN)
    const float* adj      = (const float*)d_in[1];   // (N, N)
    const float* W        = (const float*)d_in[2];   // (FOUT, FIN)
    const float* attn_src = (const float*)d_in[3];   // (H, D)
    const float* attn_dst = (const float*)d_in[4];   // (H, D)
    const float* edge_W   = (const float*)d_in[5];   // (H, 1)
    const float* edge_b   = (const float*)d_in[6];   // (H,)
    const float* fuse_W   = (const float*)d_in[7];   // (FOUT, FOUT)
    const float* fuse_b   = (const float*)d_in[8];   // (FOUT,)
    const float* ln_g     = (const float*)d_in[9];
    const float* ln_b     = (const float*)d_in[10];
    float* out = (float*)d_out;

    // Workspace layout (all offsets 256B-aligned)
    char* ws = (char*)d_ws;
    size_t off = 0;
    __bf16* xb     = (__bf16*)(ws + off); off += (size_t)NN * FIN * 2;     // 1.5 MB
    __bf16* Wb     = (__bf16*)(ws + off); off += (size_t)FOUT * FIN * 2;   // 128 KB
    __bf16* fWb    = (__bf16*)(ws + off); off += (size_t)FOUT * FOUT * 2;  // 128 KB
    float*  h      = (float*) (ws + off); off += (size_t)NN * FOUT * 4;    // 3 MB
    __bf16* hT     = (__bf16*)(ws + off); off += (size_t)FOUT * NN * 2;    // 1.5 MB
    float*  asrcT  = (float*) (ws + off); off += (size_t)NH * NN * 4;      // 96 KB
    float*  adstT  = (float*) (ws + off); off += (size_t)NH * NN * 4;      // 96 KB
    __bf16* houtb  = (__bf16*)(ws + off); off += (size_t)NN * FOUT * 2;    // 1.5 MB
    (void)ws_size; (void)in_sizes; (void)n_in; (void)out_size;

    // K1: conversions
    k_cvt_bf16<<<(NN * FIN + 255) / 256, 256, 0, stream>>>(x, xb, NN * FIN);
    k_cvt_bf16<<<(FOUT * FIN + 255) / 256, 256, 0, stream>>>(W, Wb, FOUT * FIN);
    k_cvt_bf16<<<(FOUT * FOUT + 255) / 256, 256, 0, stream>>>(fuse_W, fWb, FOUT * FOUT);

    // K2: h = x @ W^T  (and transposed bf16 copy)
    k_gemm_xw<<<NN / 16, 256, 0, stream>>>(xb, Wb, h, hT);

    // K3: attention projections
    k_attn_vec<<<(NN * NH + 255) / 256, 256, 0, stream>>>(h, attn_src, attn_dst, asrcT, adstT);

    // K4: fused masked softmax + aggregation
    k_attn_agg<<<NN / 16, 256, 0, stream>>>(adj, asrcT, adstT, edge_W, edge_b, hT, houtb);

    // K5: fuse GEMM + LayerNorm + ReLU
    k_gemm_fuse_ln<<<NN / 16, 256, 0, stream>>>(houtb, fWb, fuse_b, ln_g, ln_b, out);
}
